// FastEdgeNetwork_1726576856960
// MI455X (gfx1250) — compile-verified, matrix operand order checked by reading, not run on hardware
//
#include <hip/hip_runtime.h>
#include <hip/hip_bf16.h>

#define NN 50000     // nodes
#define NE 600000    // edges
#define DD 128       // feature dim
#define NL 3         // layers

typedef __attribute__((ext_vector_type(16))) __bf16 v16bf;
typedef __attribute__((ext_vector_type(8)))  float  v8f;
typedef __attribute__((ext_vector_type(4)))  float  v4f;
typedef __attribute__((ext_vector_type(4)))  unsigned int v4u;

// A/B fragment for V_WMMA_F32_16X16X32_BF16 (ISA 7.12.2, 16-bit 16x32 layout):
// lane L holds row (L&15); elements 0..7 = K = kbase..kbase+7, elements 8..15 =
// K = kbase+16..kbase+23, where kbase = 8*(L>>4) (+ 32*chunk handled by caller).
union FragPun { v16bf v; struct { v4u lo; v4u hi; } q; };

// Load a bf16 fragment whose two 8-element runs sit at base[0..7] and base[16..23].
template <typename P>
__device__ inline v16bf load_frag_bf16(const P* base) {
  FragPun f;
  f.q.lo = *reinterpret_cast<const v4u*>(base);
  f.q.hi = *reinterpret_cast<const v4u*>(base + 16);
  return f.v;
}

// Build the same fragment from f32 memory (rounds to bf16 in-register).
__device__ inline v16bf frag_from_f32(const float* base) {
  v4f a0 = *reinterpret_cast<const v4f*>(base);
  v4f a1 = *reinterpret_cast<const v4f*>(base + 4);
  v4f b0 = *reinterpret_cast<const v4f*>(base + 16);
  v4f b1 = *reinterpret_cast<const v4f*>(base + 20);
  v16bf r;
#pragma unroll
  for (int i = 0; i < 4; ++i) {
    r[i]      = (__bf16)a0[i];
    r[4 + i]  = (__bf16)a1[i];
    r[8 + i]  = (__bf16)b0[i];
    r[12 + i] = (__bf16)b1[i];
  }
  return r;
}

// ---------------- Kernel 1: round weights to bf16 ----------------
__global__ __launch_bounds__(256) void cvt_weights(const float* __restrict__ Wn,
                                                   const float* __restrict__ We,
                                                   __bf16* __restrict__ WnB,
                                                   __bf16* __restrict__ WeB) {
  int i = blockIdx.x * 256 + threadIdx.x;
  if (i < NL * DD * DD) {
    WnB[i] = (__bf16)Wn[i];
    WeB[i] = (__bf16)We[i];
  }
}

// ---------------- Kernel 2: H[l] = input @ W_node[l]^T ----------------
// One wave = one (16-node tile, layer). 8 waves / block.
__global__ __launch_bounds__(256) void node_gemm(const float* __restrict__ input,
                                                 const __bf16* __restrict__ WnB,
                                                 float* __restrict__ H) {
  const int lane = threadIdx.x & 31;
  const int wid  = blockIdx.x * 8 + (threadIdx.x >> 5);
  const int NT   = NN / 16;                       // 3125
  if (wid >= NT * NL) return;                     // wave-uniform exit
  const int l = wid / NT;
  const int t = wid % NT;
  const int mrow  = lane & 15;
  const int hi16  = lane >> 4;
  const int kbase = hi16 * 8;

  const float* arow = input + (size_t)(t * 16 + mrow) * DD;
  v16bf afrag[4];
#pragma unroll
  for (int c = 0; c < 4; ++c) afrag[c] = frag_from_f32(arow + c * 32 + kbase);

  const __bf16* wl = WnB + (size_t)l * DD * DD;
  float*        Hl = H   + (size_t)l * NN * DD;

#pragma unroll
  for (int nt = 0; nt < 8; ++nt) {
    const __bf16* brow = wl + (size_t)(nt * 16 + mrow) * DD + kbase;
    v8f acc = {};
#pragma unroll
    for (int c = 0; c < 4; ++c) {
      v16bf bfrag = load_frag_bf16(brow + c * 32);
      acc = __builtin_amdgcn_wmma_f32_16x16x32_bf16(false, afrag[c],
                                                    false, bfrag,
                                                    (short)0, acc, false, false);
    }
    // C layout: lane n = mrow, row = r + 8*hi16  -> coalesced 64B stores per r
#pragma unroll
    for (int r = 0; r < 8; ++r)
      Hl[(size_t)(t * 16 + r + 8 * hi16) * DD + nt * 16 + mrow] = acc[r];
  }
}

// ---------------- Kernel 3: fused 3-layer edge network ----------------
// One wave = one 16-edge x 128-feature tile, all layers in registers.
__global__ __launch_bounds__(256) void edge_fused(const float* __restrict__ eglob,
                                                  const int* __restrict__ srcs,
                                                  const __bf16* __restrict__ WeB,
                                                  const float* __restrict__ H,
                                                  float* __restrict__ out) {
  // padded rows (136 bf16 = 272B) -> conflict-free ds_load_b128 A-frag reads
  __shared__ __bf16 zshm[8][16][DD + 8];
  const int lane = threadIdx.x & 31;
  const int wib  = threadIdx.x >> 5;
  const int T    = blockIdx.x * 8 + wib;
  if (T >= NE / 16) return;                       // wave-uniform exit
  const int mrow  = lane & 15;
  const int hi16  = lane >> 4;
  const int kbase = hi16 * 8;

  // source node for each of my 8 C-layout rows (L2-resident, cacheline-shared)
  int srcid[8];
#pragma unroll
  for (int r = 0; r < 8; ++r) srcid[r] = srcs[T * 16 + r + 8 * hi16];

  // z0 = e tile in A-fragment layout (f32 -> bf16 in-register)
  const float* zrow = eglob + (size_t)(T * 16 + mrow) * DD;
  v16bf afrag[4];
#pragma unroll
  for (int c = 0; c < 4; ++c) afrag[c] = frag_from_f32(zrow + c * 32 + kbase);

  __bf16(*zl)[DD + 8] = zshm[wib];

#pragma unroll
  for (int l = 0; l < NL; ++l) {
    const __bf16* wl = WeB + (size_t)l * DD * DD;
    const float*  Hl = H   + (size_t)l * NN * DD;

#pragma unroll
    for (int nt = 0; nt < 8; ++nt) {
      const __bf16* brow = wl + (size_t)(nt * 16 + mrow) * DD + kbase;
      v8f acc = {};
#pragma unroll
      for (int c = 0; c < 4; ++c) {
        v16bf bfrag = load_frag_bf16(brow + c * 32);
        acc = __builtin_amdgcn_wmma_f32_16x16x32_bf16(false, afrag[c],
                                                      false, bfrag,
                                                      (short)0, acc, false, false);
      }
      const int n = nt * 16 + mrow;
#pragma unroll
      for (int r = 0; r < 8; ++r) {
        float h = Hl[(size_t)srcid[r] * DD + n];   // gather, coalesced over lanes 0..15
        float v = h * acc[r];
        v = v > 0.0f ? v : (__expf(v) - 1.0f);     // ELU(alpha=1)
        if (l == NL - 1) {
          out[(size_t)(T * 16 + r + 8 * hi16) * DD + n] = v;
        } else {
          zl[r + 8 * hi16][n] = (__bf16)v;         // stage next layer's z
        }
      }
    }
    if (l < NL - 1) {
      // DS ops from one wave are in-order; clobber stops compiler reordering.
      asm volatile("s_wait_dscnt 0" ::: "memory");
#pragma unroll
      for (int c = 0; c < 4; ++c)
        afrag[c] = load_frag_bf16(&zl[mrow][c * 32 + kbase]);
    }
  }
}

extern "C" void kernel_launch(void* const* d_in, const int* in_sizes, int n_in,
                              void* d_out, int out_size, void* d_ws, size_t ws_size,
                              hipStream_t stream) {
  const float* input = (const float*)d_in[0];
  const int*   esrc  = (const int*)d_in[1];       // jax x64-off -> int32
  const float* e     = (const float*)d_in[2];
  const float* Wn    = (const float*)d_in[3];
  const float* We    = (const float*)d_in[4];
  float*       out   = (float*)d_out;

  // workspace layout: H (f32, 3*50000*128 = 76.8MB) | WnB (bf16) | WeB (bf16)
  float*  H   = (float*)d_ws;
  __bf16* WnB = (__bf16*)(H + (size_t)NL * NN * DD);
  __bf16* WeB = WnB + (size_t)NL * DD * DD;

  cvt_weights<<<(NL * DD * DD + 255) / 256, 256, 0, stream>>>(Wn, We, WnB, WeB);

  const int nodeWaves = (NN / 16) * NL;           // 9375
  node_gemm<<<(nodeWaves + 7) / 8, 256, 0, stream>>>(input, WnB, H);

  const int edgeTiles = NE / 16;                  // 37500
  edge_fused<<<(edgeTiles + 7) / 8, 256, 0, stream>>>(e, esrc, WeB, H, out);
}
